// GatedBlendWM_21380347199582
// MI455X (gfx1250) — matrix operand order repaired and checked by reading, not compiled, not run on hardware
//
#include <hip/hip_runtime.h>
#include <hip/hip_bf16.h>

typedef __attribute__((ext_vector_type(16))) _Float16     v16h;
typedef __attribute__((ext_vector_type(8)))  float        v8f;
typedef __attribute__((ext_vector_type(4)))  unsigned int u32x4;
typedef __attribute__((ext_vector_type(8)))  int          i32x8;
typedef __attribute__((ext_vector_type(4)))  int          i32x4;

#define B_    32
#define C_    8
#define HID_  128
#define HW_   256
#define NPIX_ (B_ * C_ * HW_ * HW_)
#define R_    3.9f
#define EPS_  0.3f
#define BETA_ 0.15f

// Packed-weight region sizes (f16 elements), laid out contiguously in ws:
//   WpB (8nt*3kc*32lane*16) | WuB (4kc*32*16) | Wg1B (7kc*32*16) | Wg2B (32*16)
#define WPB_N  (8 * 3 * 32 * 16)   // 12288
#define WUB_N  (4 * 32 * 16)       // 2048
#define WG1B_N (7 * 32 * 16)       // 3584
#define WG2B_N (32 * 16)           // 512

// im2col LDS offsets as compile-time constants (k is always a literal after
// unrolling; the hi-half choice is a single v_cndmask).  k >= valid-K maps to
// a zeroed tail row so padded taps need no branch.
#define NCA_ZOFS  (8 * 3 * 34)      // 816
#define GATE_ZOFS (24 * 3 * 34)     // 2448
__host__ __device__ constexpr int nca_kofs(int k) {
    return (k < 72) ? ((k / 9) * 102 + ((k % 9) / 3) * 34 + (k % 9) % 3) : NCA_ZOFS;
}
__host__ __device__ constexpr int gate_kofs(int k) {
    return (k < 216) ? ((k / 9) * 102 + ((k % 9) / 3) * 34 + (k % 9) % 3) : GATE_ZOFS;
}

// ---------------------------------------------------------------------------
// Tensor Data Mover: 1-D DMA of `nelem` f16 from global to LDS.
// D# per CDNA5 ISA; tracked by TENSORcnt.
// ---------------------------------------------------------------------------
__device__ __forceinline__ void tdm_load_f16_1d(const void* gptr,
                                                unsigned lds_byte_off,
                                                unsigned nelem) {
    unsigned long long ga = (unsigned long long)(uintptr_t)gptr;
    u32x4 g0;
    g0[0] = 1u;                                            // count=1 (valid, user)
    g0[1] = lds_byte_off;                                  // lds_addr
    g0[2] = (unsigned)(ga & 0xFFFFFFFFu);                  // global_addr[31:0]
    g0[3] = (unsigned)((ga >> 32) & 0x01FFFFFFu)           // global_addr[56:32]
          | (2u << 30);                                    // type=2 ("image")
    i32x8 g1;
    g1[0] = (int)(1u << 16);                               // data_size=2 bytes
    g1[1] = (int)((nelem & 0xFFFFu) << 16);                // tensor_dim0[15:0]
    g1[2] = (int)((nelem >> 16) | (1u << 16));             // tensor_dim0 hi, tensor_dim1=1
    g1[3] = (int)((nelem & 0xFFFFu) << 16);                // tile_dim0
    g1[4] = 1;                                             // tile_dim1=1
    g1[5] = (int)nelem;                                    // tensor_dim0_stride
    g1[6] = 0;
    g1[7] = 0;
    i32x4 z4 = {0, 0, 0, 0};
    i32x8 z8 = {0, 0, 0, 0, 0, 0, 0, 0};
    __builtin_amdgcn_tensor_load_to_lds(g0, g1, z4, z4, z8, 0);
}

__device__ __forceinline__ unsigned lds_off_of(const void* p) {
    return (unsigned)(unsigned long long)(uintptr_t)p;     // low 32b = LDS offset
}

// ---------------------------------------------------------------------------
// Pack B-matrices into per-lane v16h order for v_wmma_f32_16x16x32_f16.
// ---------------------------------------------------------------------------
__global__ void pack_weights_kernel(const float* __restrict__ Wp,
                                    const float* __restrict__ Wu,
                                    const float* __restrict__ Wg1,
                                    const float* __restrict__ Wg2,
                                    _Float16* __restrict__ WB) {
    int tid = threadIdx.x;
    _Float16* WpB  = WB;
    _Float16* WuB  = WpB + WPB_N;
    _Float16* Wg1B = WuB + WUB_N;
    _Float16* Wg2B = Wg1B + WG1B_N;
    for (int i = tid; i < WPB_N; i += 256) {               // K = c*9+ky*3+kx (72 pad 96)
        int j = i & 15, lane = (i >> 4) & 31;
        int kc = (i >> 9) % 3, nt = (i >> 9) / 3;
        int k = kc * 32 + ((lane >= 16) ? 8 : 0) + j + ((j >= 8) ? 8 : 0);
        int n = nt * 16 + (lane & 15);
        float v = 0.f;
        if (k < 72) { int c = k / 9, r = k % 9; v = Wp[(n * 8 + c) * 9 + r]; }
        WpB[i] = (_Float16)v;
    }
    for (int i = tid; i < WUB_N; i += 256) {               // K=0..127, N pad 8->16
        int j = i & 15, lane = (i >> 4) & 31, kc = i >> 9;
        int k = kc * 32 + ((lane >= 16) ? 8 : 0) + j + ((j >= 8) ? 8 : 0);
        int n = lane & 15;
        WuB[i] = (_Float16)((n < 8) ? Wu[n * 128 + k] : 0.f);
    }
    for (int i = tid; i < WG1B_N; i += 256) {              // K over 24ch (216 pad 224)
        int j = i & 15, lane = (i >> 4) & 31, kc = i >> 9;
        int k = kc * 32 + ((lane >= 16) ? 8 : 0) + j + ((j >= 8) ? 8 : 0);
        int n = lane & 15;
        float v = 0.f;
        if (n < 8 && k < 216) { int c = k / 9, r = k % 9; v = Wg1[(n * 24 + c) * 9 + r]; }
        Wg1B[i] = (_Float16)v;
    }
    for (int i = tid; i < WG2B_N; i += 256) {              // K=0..7 pad 32
        int j = i & 15, lane = (i >> 4) & 31;
        int k = ((lane >= 16) ? 8 : 0) + j + ((j >= 8) ? 8 : 0);
        int n = lane & 15;
        Wg2B[i] = (_Float16)((n < 8 && k < 8) ? Wg2[n * 8 + k] : 0.f);
    }
}

// ---------------------------------------------------------------------------
// One CML step (VALU-bound, L2-resident state): 32x32 tile + halo in LDS.
// ---------------------------------------------------------------------------
__global__ void cml_step_kernel(const float* __restrict__ g,
                                const float* __restrict__ drv,
                                const float* __restrict__ Kl,
                                float* __restrict__ gout) {
    __shared__ float sm[34][36];
    __shared__ float kk[9];
    int bc = blockIdx.z;
    int c  = bc & 7;
    int tx0 = blockIdx.x * 32, ty0 = blockIdx.y * 32;
    int tid = threadIdx.x;
    if (tid < 9) kk[tid] = Kl[c * 9 + tid];
    const float* gp = g + ((size_t)bc << 16);
    for (int i = tid; i < 34 * 34; i += 256) {
        int ly = i / 34, lx = i % 34;
        int yy = ty0 + ly - 1, xx = tx0 + lx - 1;
        float v = 0.f;
        if ((unsigned)yy < 256u && (unsigned)xx < 256u) v = gp[yy * 256 + xx];
        sm[ly][lx] = R_ * v * (1.f - v);                   // mapped(0)=0 == zero pad
    }
    __syncthreads();
    const float* dp = drv + ((size_t)bc << 16);
    float* op = gout + ((size_t)bc << 16);
    for (int i = tid; i < 1024; i += 256) {
        int ly = i >> 5, lx = i & 31;
        float loc =
              kk[0] * sm[ly][lx]     + kk[1] * sm[ly][lx + 1]     + kk[2] * sm[ly][lx + 2]
            + kk[3] * sm[ly + 1][lx] + kk[4] * sm[ly + 1][lx + 1] + kk[5] * sm[ly + 1][lx + 2]
            + kk[6] * sm[ly + 2][lx] + kk[7] * sm[ly + 2][lx + 1] + kk[8] * sm[ly + 2][lx + 2];
        float phys = (1.f - EPS_) * sm[ly + 1][lx + 1] + EPS_ * loc;
        int yy = ty0 + ly, xx = tx0 + lx;
        op[yy * 256 + xx] = (1.f - BETA_) * phys + BETA_ * dp[yy * 256 + xx];
    }
}

// ---------------------------------------------------------------------------
// NCA branch. Block = 128 thr (4 waves) = 32 consecutive pixels of one row.
// TDM pulls packed weights (28 KB) into LDS while waves stage the patch.
// GEMM1: M=32,K=96,N=128 (wave: 2 n-tiles x 2 m-tiles, B reused across m).
// GEMM2: M=32,K=128,N=16 on waves 0/1.
// ---------------------------------------------------------------------------
__global__ void nca_kernel(const float* __restrict__ x,
                           const _Float16* __restrict__ WpB,   // WuB = WpB+WPB_N
                           const float* __restrict__ bp,
                           const float* __restrict__ bu,
                           float* __restrict__ nca) {
    __shared__ __align__(32) _Float16 wlds[WPB_N + WUB_N];     // 28 KB weights
    __shared__ _Float16 patch[NCA_ZOFS + 34];                  // rows + zero tail
    __shared__ _Float16 htile[32][136];
    int x0 = blockIdx.x * 32, y = blockIdx.y, b = blockIdx.z;
    int tid = threadIdx.x;
    int wave = tid >> 5, lane = tid & 31;
    int m = lane & 15, hi = lane >> 4;

    if (tid < 32)                                              // one wave issues DMA
        tdm_load_f16_1d(WpB, lds_off_of(wlds), WPB_N + WUB_N);

    for (int i = tid; i < 34; i += 128) patch[NCA_ZOFS + i] = (_Float16)0.f;
    for (int i = tid; i < 8 * 3 * 34; i += 128) {              // stage x patch
        int c = i / 102, rem = i % 102, ky = rem / 34, ix = rem % 34;
        int yy = y + ky - 1, xx = x0 + ix - 1;
        float v = 0.f;
        if ((unsigned)yy < 256u && (unsigned)xx < 256u)
            v = x[(((size_t)(b * 8 + c) * 256) + yy) * 256 + xx];
        patch[i] = (_Float16)v;
    }
    __builtin_prefetch(&x[(((size_t)(b * 8) * 256) + y) * 256 + ((x0 + 32) & 255)], 0, 1);
    if (tid < 32) __builtin_amdgcn_s_wait_tensorcnt(0);
    __syncthreads();

    v8f acc[2][2];                                             // [ntile][mtile]
    #pragma unroll
    for (int t = 0; t < 2; ++t)
        #pragma unroll
        for (int mt = 0; mt < 2; ++mt)
            acc[t][mt] = (v8f){0.f, 0.f, 0.f, 0.f, 0.f, 0.f, 0.f, 0.f};

    #pragma unroll
    for (int kc = 0; kc < 3; ++kc) {
        v16h a0, a1;                                           // m-tiles 0 and 1
        #pragma unroll
        for (int j = 0; j < 8; ++j) {
            // k(lo half) = kc*32 + hi*8 + j ; k(hi half) = +16. Offsets are
            // compile-time for both hi cases; runtime cost = 1 select + add.
            int o1 = hi ? nca_kofs(kc * 32 + 8 + j)  : nca_kofs(kc * 32 + j);
            int o2 = hi ? nca_kofs(kc * 32 + 24 + j) : nca_kofs(kc * 32 + 16 + j);
            a0[j]     = patch[o1 + m];
            a0[j + 8] = patch[o2 + m];
            a1[j]     = patch[o1 + 16 + m];
            a1[j + 8] = patch[o2 + 16 + m];
        }
        #pragma unroll
        for (int t = 0; t < 2; ++t) {
            int ntile = wave * 2 + t;
            v16h bf = *(const v16h*)(wlds + (size_t)((ntile * 3 + kc) * 32 + lane) * 16);
            acc[t][0] = __builtin_amdgcn_wmma_f32_16x16x32_f16(false, a0, false, bf,
                                                               (short)0, acc[t][0], false, false);
            acc[t][1] = __builtin_amdgcn_wmma_f32_16x16x32_f16(false, a1, false, bf,
                                                               (short)0, acc[t][1], false, false);
        }
    }
    #pragma unroll
    for (int t = 0; t < 2; ++t) {
        int n = (wave * 2 + t) * 16 + m;
        float bias = bp[n];
        #pragma unroll
        for (int mt = 0; mt < 2; ++mt)
            #pragma unroll
            for (int r = 0; r < 8; ++r) {
                float hv = acc[t][mt][r] + bias;
                hv = hv > 0.f ? hv : 0.f;
                htile[mt * 16 + r + hi * 8][n] = (_Float16)hv;
            }
    }
    __syncthreads();

    if (wave < 2) {                                            // GEMM2: one m-tile per wave
        v8f a2 = {0.f, 0.f, 0.f, 0.f, 0.f, 0.f, 0.f, 0.f};
        const _Float16* wu = wlds + WPB_N;
        const _Float16* hr = &htile[wave * 16 + m][0];
        #pragma unroll
        for (int kc = 0; kc < 4; ++kc) {
            v16h a;
            #pragma unroll
            for (int j = 0; j < 8; ++j) {
                a[j]     = hr[kc * 32 + hi * 8 + j];
                a[j + 8] = hr[kc * 32 + hi * 8 + 16 + j];
            }
            v16h bf = *(const v16h*)(wu + (size_t)(kc * 32 + lane) * 16);
            a2 = __builtin_amdgcn_wmma_f32_16x16x32_f16(false, a, false, bf,
                                                        (short)0, a2, false, false);
        }
        int n = lane & 15;
        if (n < 8) {
            float bias = bu[n];
            #pragma unroll
            for (int r = 0; r < 8; ++r) {
                float v = a2[r] + bias;
                v = 1.f / (1.f + __expf(-v));
                nca[(((size_t)(b * 8 + n) * 256) + y) * 256 + (x0 + wave * 16 + r + hi * 8)] = v;
            }
        }
    }
}

// ---------------------------------------------------------------------------
// Gate + blend. Block = 64 thr (2 waves) = 32 pixels; each wave owns one
// 16-pixel m-tile for both WMMA convs.  Gate weights arrive via TDM.
// ---------------------------------------------------------------------------
__global__ void gate_kernel(const float* __restrict__ x,
                            const float* __restrict__ cml,
                            const float* __restrict__ nca,
                            const _Float16* __restrict__ Wg1B,  // Wg2B = Wg1B+WG1B_N
                            const float* __restrict__ bg1,
                            const float* __restrict__ bg2,
                            float* __restrict__ out) {
    __shared__ __align__(32) _Float16 wlds[WG1B_N + WG2B_N];    // 8 KB
    __shared__ _Float16 patch[GATE_ZOFS + 34];                  // rows + zero tail
    __shared__ _Float16 ghid[32][32];
    int x0 = blockIdx.x * 32, y = blockIdx.y, b = blockIdx.z;
    int tid = threadIdx.x, wave = tid >> 5, lane = tid & 31;
    int m = lane & 15, hi = lane >> 4;

    if (tid < 32)
        tdm_load_f16_1d(Wg1B, lds_off_of(wlds), WG1B_N + WG2B_N);

    for (int i = tid; i < 32 * 32; i += 64) ((_Float16*)ghid)[i] = (_Float16)0.f;
    for (int i = tid; i < 34; i += 64) patch[GATE_ZOFS + i] = (_Float16)0.f;
    for (int i = tid; i < 24 * 3 * 34; i += 64) {
        int c = i / 102, rem = i % 102, ky = rem / 34, ix = rem % 34;
        int yy = y + ky - 1, xx = x0 + ix - 1;
        float v = 0.f;
        if ((unsigned)yy < 256u && (unsigned)xx < 256u) {
            if (c < 8) {
                v = x[(((size_t)(b * 8 + c) * 256) + yy) * 256 + xx];
            } else if (c < 16) {
                v = cml[(((size_t)(b * 8 + (c - 8)) * 256) + yy) * 256 + xx];
                v = fminf(fmaxf(v, 0.0001f), 0.9999f);
            } else {
                v = nca[(((size_t)(b * 8 + (c - 16)) * 256) + yy) * 256 + xx];
            }
        }
        patch[i] = (_Float16)v;
    }
    if (tid < 32) __builtin_amdgcn_s_wait_tensorcnt(0);
    __syncthreads();

    // conv1 (3x3, 24->8): K = 216 pad 224, m-tile = wave
    int col = wave * 16 + m;
    v8f acc = {0.f, 0.f, 0.f, 0.f, 0.f, 0.f, 0.f, 0.f};
    #pragma unroll
    for (int kc = 0; kc < 7; ++kc) {
        v16h a;
        #pragma unroll
        for (int j = 0; j < 8; ++j) {
            int o1 = hi ? gate_kofs(kc * 32 + 8 + j)  : gate_kofs(kc * 32 + j);
            int o2 = hi ? gate_kofs(kc * 32 + 24 + j) : gate_kofs(kc * 32 + 16 + j);
            a[j]     = patch[o1 + col];
            a[j + 8] = patch[o2 + col];
        }
        v16h bf = *(const v16h*)(wlds + (size_t)(kc * 32 + lane) * 16);
        acc = __builtin_amdgcn_wmma_f32_16x16x32_f16(false, a, false, bf,
                                                     (short)0, acc, false, false);
    }
    {
        int n = lane & 15;
        if (n < 8) {
            float bias = bg1[n];
            #pragma unroll
            for (int r = 0; r < 8; ++r) {
                float v = acc[r] + bias;
                ghid[wave * 16 + r + hi * 8][n] = (_Float16)(v > 0.f ? v : 0.f);
            }
        }
    }
    __syncthreads();

    // conv2 (1x1, 8->8): K = 8 pad 32, then sigmoid + blend
    {
        v16h a;
        #pragma unroll
        for (int j = 0; j < 8; ++j) {
            a[j]     = ghid[wave * 16 + m][hi * 8 + j];
            a[j + 8] = ghid[wave * 16 + m][hi * 8 + 16 + j];
        }
        v8f a2 = {0.f, 0.f, 0.f, 0.f, 0.f, 0.f, 0.f, 0.f};
        v16h bf = *(const v16h*)(wlds + WG1B_N + (size_t)lane * 16);
        a2 = __builtin_amdgcn_wmma_f32_16x16x32_f16(false, a, false, bf,
                                                    (short)0, a2, false, false);
        int n = lane & 15;
        if (n < 8) {
            float bias = bg2[n];
            #pragma unroll
            for (int r = 0; r < 8; ++r) {
                int mm = wave * 16 + r + hi * 8;
                size_t idx = (((size_t)(b * 8 + n) * 256) + y) * 256 + (x0 + mm);
                float gv = 1.f / (1.f + __expf(-(a2[r] + bias)));
                float cv = fminf(fmaxf(cml[idx], 0.0001f), 0.9999f);
                out[idx] = gv * cv + (1.f - gv) * nca[idx];
            }
        }
    }
}

// ---------------------------------------------------------------------------
extern "C" void kernel_launch(void* const* d_in, const int* in_sizes, int n_in,
                              void* d_out, int out_size, void* d_ws, size_t ws_size,
                              hipStream_t stream) {
    (void)in_sizes; (void)n_in; (void)out_size; (void)ws_size;
    const float* x   = (const float*)d_in[0];
    const float* Kl  = (const float*)d_in[1];
    const float* Wp  = (const float*)d_in[2];
    const float* bp  = (const float*)d_in[3];
    const float* Wu  = (const float*)d_in[4];
    const float* bu  = (const float*)d_in[5];
    const float* Wg1 = (const float*)d_in[6];
    const float* bg1 = (const float*)d_in[7];
    const float* Wg2 = (const float*)d_in[8];
    const float* bg2 = (const float*)d_in[9];
    float* out = (float*)d_out;

    float* gridA = (float*)d_ws;
    float* gridB = gridA + NPIX_;
    float* nca   = gridB + NPIX_;
    _Float16* WB   = (_Float16*)(nca + NPIX_);      // WpB|WuB|Wg1B|Wg2B contiguous
    _Float16* Wg1B = WB + WPB_N + WUB_N;

    pack_weights_kernel<<<1, 256, 0, stream>>>(Wp, Wu, Wg1, Wg2, WB);

    const float* src = x;
    float* dst = gridA;
    for (int s = 0; s < 15; ++s) {
        cml_step_kernel<<<dim3(8, 8, B_ * C_), 256, 0, stream>>>(src, x, Kl, dst);
        src = dst;
        dst = (dst == gridA) ? gridB : gridA;
    }
    const float* cml = src;

    nca_kernel<<<dim3(8, 256, B_), 128, 0, stream>>>(x, WB, bp, bu, nca);

    gate_kernel<<<dim3(8, 256, B_), 64, 0, stream>>>(x, cml, nca,
                                                     Wg1B, bg1, bg2, out);
}
